// ReversibleSelfAttention_76579266888328
// MI455X (gfx1250) — matrix operand order repaired and checked by reading, not compiled, not run on hardware
//
#include <hip/hip_runtime.h>
#include <stdint.h>

// ---------------------------------------------------------------------------
// ReversibleSelfAttention for MI455X (gfx1250, wave32, WMMA bf16 + TDM)
//   x:[2,2048,2048] f32; w_q/k/v/o:[2048,2048] f32; gamma/beta:[2048]
//   out = x + Attn(LN(x))
// bf16 WMMA (16x16x32) with fp32 accumulate; flash-style online softmax;
// Tensor Data Mover stages V tiles into LDS in the attention kernel.
// ---------------------------------------------------------------------------

#define D_MODEL 2048
#define N_HEADS 16
#define D_K     128
#define SEQ     2048
#define BATCH   2
#define TOK     (BATCH * SEQ)  // 4096
#define LN_EPS  1e-5f

typedef __attribute__((ext_vector_type(16))) __bf16 bf16x16;
typedef __attribute__((ext_vector_type(8)))  float  f32x8;
typedef __attribute__((ext_vector_type(4)))  unsigned int u32x4;
typedef __attribute__((ext_vector_type(8)))  int    i32x8;
typedef __attribute__((ext_vector_type(4)))  int    i32x4;

union FragB {
    bf16x16 v;
    uint4 q[2];
    unsigned short h[16];
};

__device__ __forceinline__ unsigned short f2bf(float f) {
    unsigned int u = __float_as_uint(f);
    unsigned int r = (u + 0x7FFFu + ((u >> 16) & 1u)) >> 16;  // RNE
    return (unsigned short)r;
}

__device__ __forceinline__ f32x8 zero8() {
    f32x8 z;
#pragma unroll
    for (int i = 0; i < 8; ++i) z[i] = 0.0f;
    return z;
}

// A fragment (16x32, bf16): lane l -> row l%16, halves [kb, kb+8) and
// [kb+16, kb+24), kb = (l<16 ? 0 : 8). Row-major source, stride `ld` halves.
__device__ __forceinline__ bf16x16 load_a_frag(const unsigned short* base, int ld,
                                               int row, int k, int lane) {
    FragB f;
    const unsigned short* p = base + (size_t)row * ld + k + ((lane & 16) ? 8 : 0);
    f.q[0] = *(const uint4*)(p);
    f.q[1] = *(const uint4*)(p + 16);
    return f.v;
}

// B fragment (32x16, bf16): lane l -> column l%16, halves [kb, kb+8) and
// [kb+16, kb+24), kb = (l<16 ? 0 : 8). Column n of B is contiguous along K at row `col_row`
// of a row-major [N, K] array (e.g. W for y = x*W^T).
__device__ __forceinline__ bf16x16 load_b_frag(const unsigned short* base, int ld,
                                               int col_row, int k, int lane) {
    FragB f;
    const unsigned short* p = base + (size_t)col_row * ld + k + ((lane & 16) ? 8 : 0);
    f.q[0] = *(const uint4*)(p);
    f.q[1] = *(const uint4*)(p + 16);
    return f.v;
}

__device__ __forceinline__ f32x8 wmma_bf16(bf16x16 a, bf16x16 b, f32x8 c) {
    return __builtin_amdgcn_wmma_f32_16x16x32_bf16(false, a, false, b, (short)0, c,
                                                   false, false);
}

// ---------------------------------------------------------------------------
// Kernel 1: LayerNorm + cast to bf16.  One block per token row.
// ---------------------------------------------------------------------------
__global__ __launch_bounds__(256) void ln_kernel(const float* __restrict__ x,
                                                 const float* __restrict__ gamma,
                                                 const float* __restrict__ beta,
                                                 unsigned short* __restrict__ nx) {
    const int tokr = blockIdx.x;
    const int tid = threadIdx.x;
    const float* row = x + (size_t)tokr * D_MODEL;

    float s = 0.f, ss = 0.f;
    float vals[8];
#pragma unroll
    for (int i = 0; i < 8; ++i) {
        float v = row[tid + i * 256];
        vals[i] = v;
        s += v;
        ss += v * v;
    }
#pragma unroll
    for (int off = 16; off; off >>= 1) {
        s += __shfl_xor(s, off, 32);
        ss += __shfl_xor(ss, off, 32);
    }
    __shared__ float rs[8], rss[8];
    const int wave = tid >> 5, lane = tid & 31;
    if (lane == 0) { rs[wave] = s; rss[wave] = ss; }
    __syncthreads();
    s = 0.f; ss = 0.f;
#pragma unroll
    for (int i = 0; i < 8; ++i) { s += rs[i]; ss += rss[i]; }
    const float mu = s * (1.0f / D_MODEL);
    const float var = ss * (1.0f / D_MODEL) - mu * mu;
    const float rstd = rsqrtf(var + LN_EPS);
#pragma unroll
    for (int i = 0; i < 8; ++i) {
        const int c = tid + i * 256;
        const float v = (vals[i] - mu) * rstd * gamma[c] + beta[c];
        nx[(size_t)tokr * D_MODEL + c] = f2bf(v);
    }
}

// ---------------------------------------------------------------------------
// Kernel 2: cast the four weight matrices to bf16.
// ---------------------------------------------------------------------------
__global__ __launch_bounds__(256) void cast_w_kernel(
    const float* __restrict__ w0, const float* __restrict__ w1,
    const float* __restrict__ w2, const float* __restrict__ w3,
    unsigned short* __restrict__ o0, unsigned short* __restrict__ o1,
    unsigned short* __restrict__ o2, unsigned short* __restrict__ o3) {
    const float* src;
    unsigned short* dst;
    switch (blockIdx.y) {
        case 0: src = w0; dst = o0; break;
        case 1: src = w1; dst = o1; break;
        case 2: src = w2; dst = o2; break;
        default: src = w3; dst = o3; break;
    }
    const size_t i = ((size_t)blockIdx.x * 256 + threadIdx.x) * 4;
    const float4 v = *(const float4*)(src + i);
    dst[i + 0] = f2bf(v.x);
    dst[i + 1] = f2bf(v.y);
    dst[i + 2] = f2bf(v.z);
    dst[i + 3] = f2bf(v.w);
}

// ---------------------------------------------------------------------------
// Kernel 3: fused QKV projection GEMM.  y = nx * W^T, head-split bf16
// [B,H,S,DK] output.  Block = 256 thr (8 waves); wave tile 32x128 (two A
// fragments share every B fragment -> 16 WMMA per 20 b128 loads per k-step).
// ---------------------------------------------------------------------------
__global__ __launch_bounds__(256) void qkv_gemm_kernel(
    const unsigned short* __restrict__ nx, const unsigned short* __restrict__ wq,
    const unsigned short* __restrict__ wk, const unsigned short* __restrict__ wv,
    unsigned short* __restrict__ qb, unsigned short* __restrict__ kb,
    unsigned short* __restrict__ vb) {
    const unsigned short* w;
    unsigned short* dst;
    if (blockIdx.z == 0) { w = wq; dst = qb; }
    else if (blockIdx.z == 1) { w = wk; dst = kb; }
    else { w = wv; dst = vb; }

    const int lane = threadIdx.x & 31;
    const int wave = threadIdx.x >> 5;
    const int m0 = blockIdx.x * 256 + wave * 32;
    const int n0 = blockIdx.y * 128;

    f32x8 acc0[8], acc1[8];
#pragma unroll
    for (int i = 0; i < 8; ++i) { acc0[i] = zero8(); acc1[i] = zero8(); }

    const int arow = m0 + (lane & 15);
    for (int k = 0; k < D_MODEL; k += 32) {
        const bf16x16 a0 = load_a_frag(nx, D_MODEL, arow, k, lane);
        const bf16x16 a1 = load_a_frag(nx, D_MODEL, arow + 16, k, lane);
#pragma unroll
        for (int nt = 0; nt < 8; ++nt) {
            const bf16x16 b = load_b_frag(w, D_MODEL, n0 + nt * 16 + (lane & 15), k, lane);
            acc0[nt] = wmma_bf16(a0, b, acc0[nt]);
            acc1[nt] = wmma_bf16(a1, b, acc1[nt]);
        }
    }

    const int roff = (lane & 16) ? 8 : 0;
#pragma unroll
    for (int half = 0; half < 2; ++half) {
        const int rowbase = m0 + half * 16 + roff;
#pragma unroll
        for (int nt = 0; nt < 8; ++nt) {
            const int e = n0 + nt * 16 + (lane & 15);
            const int h = e >> 7, dk = e & 127;
#pragma unroll
            for (int r = 0; r < 8; ++r) {
                const int tokr = rowbase + r;
                const int bb = tokr >> 11, srow = tokr & (SEQ - 1);
                const size_t idx = ((((size_t)bb * N_HEADS + h) * SEQ) + srow) * D_K + dk;
                const float vvv = half ? acc1[nt][r] : acc0[nt][r];
                dst[idx] = f2bf(vvv);
            }
        }
    }
}

// ---------------------------------------------------------------------------
// TDM: issue a 2-D tensor load of a [32 keys x 128 dk] bf16 tile into LDS.
// Descriptor per CDNA5 ISA (groups 2/3 NULL, 2-D).
// This toolchain exposes the 6-arg builtin form.
// ---------------------------------------------------------------------------
#if defined(__gfx1250__) && __has_builtin(__builtin_amdgcn_tensor_load_to_lds)
#define HAVE_TDM 1
__device__ __forceinline__ void tdm_load_v_tile(const unsigned short* gptr,
                                                unsigned lds_byte_off) {
    const unsigned long long ga = (unsigned long long)(uintptr_t)gptr;
    u32x4 g0;
    g0[0] = 1u;                                   // count=1 valid, is_restore=0
    g0[1] = lds_byte_off;                         // lds_addr
    g0[2] = (unsigned)(ga & 0xFFFFFFFFull);       // global_addr[31:0]
    g0[3] = (unsigned)((ga >> 32) & 0x01FFFFFFull) | (2u << 30);  // [56:32]|type=2
    i32x8 g1;
    g1[0] = (1 << 16);              // workgroup_mask=0, data_size=1 (2 bytes)
    g1[1] = (int)((unsigned)D_K << 16);   // tensor_dim0[15:0] = 128 (bits 63:48)
    g1[2] = (int)((unsigned)SEQ << 16);   // dim0 hi=0; tensor_dim1[15:0]=2048 (95:80)
    g1[3] = (int)(128u << 16);            // dim1 hi=0; tile_dim0=128 (127:112)
    g1[4] = 32;                           // tile_dim1=32 keys; tile_dim2=0
    g1[5] = D_K;                          // tensor_dim0_stride = 128 (191:160)
    g1[6] = 0;                            // stride hi, tensor_dim1_stride lo
    g1[7] = 0;
    const i32x4 z4 = {0, 0, 0, 0};
    const i32x8 z8 = {0, 0, 0, 0, 0, 0, 0, 0};
    __builtin_amdgcn_tensor_load_to_lds(g0, g1, z4, z4, z8, 0);
}
#else
#define HAVE_TDM 0
#endif

// ---------------------------------------------------------------------------
// Kernel 4: flash attention.  Block = 256 thr (8 waves), one (b,h), 128-row q
// tile.  scores^T = K*Q^T so softmax reduction is in-lane + shfl_xor(16);
// P transposed C->A layout through per-wave LDS; V tiles staged by the TDM
// then transposed to VT[dk][key] for the ctx WMMA B operands.
// ---------------------------------------------------------------------------
__global__ __launch_bounds__(256) void attn_kernel(
    const unsigned short* __restrict__ qbuf, const unsigned short* __restrict__ kbuf,
    const unsigned short* __restrict__ vbuf, unsigned short* __restrict__ ctx) {
    __shared__ __align__(16) unsigned short Vstage[32][128];  // [key][dk] 8 KB
    __shared__ __align__(16) unsigned short VT[128][32];      // [dk][key] 8 KB
    __shared__ __align__(16) unsigned short P[8][16][32];     // [wave][q][key] 8 KB

    const int bh = blockIdx.x;          // b*16 + h
    const int q0 = blockIdx.y * 128;
    const int tid = threadIdx.x, lane = tid & 31, wave = tid >> 5;
    const int qc = lane & 15;
    const int roff = (lane & 16) ? 8 : 0;

    const unsigned short* qh = qbuf + (size_t)bh * SEQ * D_K;
    const unsigned short* kh = kbuf + (size_t)bh * SEQ * D_K;
    const unsigned short* vh = vbuf + (size_t)bh * SEQ * D_K;

    // Preload Q as B-fragments (column q, K = dk), 4 chunks of 32.
    bf16x16 qfr[4];
    const int qrow = q0 + wave * 16 + qc;
#pragma unroll
    for (int kk = 0; kk < 4; ++kk) qfr[kk] = load_b_frag(qh, D_K, qrow, kk * 32, lane);

    f32x8 ctxacc[8];
#pragma unroll
    for (int i = 0; i < 8; ++i) ctxacc[i] = zero8();

    float m = -3.4e38f, lsum = 0.f;
    const float scale = 0.08838834764831845f;  // 1/sqrt(128)

#if HAVE_TDM
    const unsigned vstage_off = (unsigned)(uintptr_t)(&Vstage[0][0]);
#endif

    for (int j = 0; j < SEQ; j += 32) {
        // prefetch next K/V tiles while we work on this one
        if (j + 32 < SEQ) {
            __builtin_prefetch(kh + (size_t)(j + 32) * D_K, 0, 1);
            __builtin_prefetch(vh + (size_t)(j + 32) * D_K, 0, 1);
        }

#if HAVE_TDM
        // ---- TDM: V tile [32 x 128] -> Vstage (row-major) ----
        if (wave == 0) {
            tdm_load_v_tile(vh + (size_t)j * D_K, vstage_off);
            __builtin_amdgcn_s_wait_tensorcnt(0);
        }
        __syncthreads();
        // ---- transpose Vstage -> VT[dk][key] ----
        {
            const int kr = tid >> 3;          // key 0..31
            const int dkb = (tid & 7) * 16;   // dk base
            unsigned short tmp[16];
            *(uint4*)(tmp) = *(const uint4*)(&Vstage[kr][dkb]);
            *(uint4*)(tmp + 8) = *(const uint4*)(&Vstage[kr][dkb + 8]);
#pragma unroll
            for (int i = 0; i < 16; ++i) VT[dkb + i][kr] = tmp[i];
        }
#else
        // ---- fallback: stage V^T directly from global ----
        {
            const int kr = tid >> 3;
            const int dkb = (tid & 7) * 16;
            const unsigned short* vp = vh + (size_t)(j + kr) * D_K + dkb;
            unsigned short tmp[16];
            *(uint4*)(tmp) = *(const uint4*)(vp);
            *(uint4*)(tmp + 8) = *(const uint4*)(vp + 8);
#pragma unroll
            for (int i = 0; i < 16; ++i) VT[dkb + i][kr] = tmp[i];
        }
#endif
        __syncthreads();

        // ---- scores^T (two 16x16 tiles: keys x q) ----
        f32x8 sacc[2];
        sacc[0] = zero8();
        sacc[1] = zero8();
#pragma unroll
        for (int kk = 0; kk < 4; ++kk) {
#pragma unroll
            for (int t = 0; t < 2; ++t) {
                const bf16x16 kf = load_a_frag(kh, D_K, j + t * 16 + qc, kk * 32, lane);
                sacc[t] = wmma_bf16(kf, qfr[kk], sacc[t]);
            }
        }

        // ---- online softmax (per lane: column q = qc, 16 key elements) ----
        float sv[2][8];
        float mt = -3.4e38f;
#pragma unroll
        for (int t = 0; t < 2; ++t)
#pragma unroll
            for (int r = 0; r < 8; ++r) {
                sv[t][r] = sacc[t][r] * scale;
                mt = fmaxf(mt, sv[t][r]);
            }
        mt = fmaxf(mt, __shfl_xor(mt, 16, 32));
        const float mnew = fmaxf(m, mt);
        const float alpha = __expf(m - mnew);
        m = mnew;

        float p[2][8];
        float ls = 0.f;
#pragma unroll
        for (int t = 0; t < 2; ++t)
#pragma unroll
            for (int r = 0; r < 8; ++r) {
                p[t][r] = __expf(sv[t][r] - mnew);
                ls += p[t][r];
            }
        ls += __shfl_xor(ls, 16, 32);
        lsum = lsum * alpha + ls;

        // rescale ctx accumulators by per-q alpha (alpha lives in lane q)
#pragma unroll
        for (int r = 0; r < 8; ++r) {
            const float ar = __shfl(alpha, r + roff, 32);
#pragma unroll
            for (int nt = 0; nt < 8; ++nt) ctxacc[nt][r] *= ar;
        }

        // ---- write P (keys x q, C layout) to LDS as P[q][key] ----
#pragma unroll
        for (int t = 0; t < 2; ++t)
#pragma unroll
            for (int r = 0; r < 8; ++r)
                P[wave][qc][t * 16 + roff + r] = f2bf(p[t][r]);

        // ---- reload P as an A fragment (row q, K = keys) ----
        FragB pf;
        {
            const unsigned short* pp = &P[wave][qc][(lane & 16) ? 8 : 0];
            pf.q[0] = *(const uint4*)(pp);
            pf.q[1] = *(const uint4*)(pp + 16);
        }

        // ---- ctx += P(16x32) * V(32x128) : B fragments from VT in LDS ----
#pragma unroll
        for (int nt = 0; nt < 8; ++nt) {
            FragB vf;
            const unsigned short* vp2 = &VT[nt * 16 + qc][(lane & 16) ? 8 : 0];
            vf.q[0] = *(const uint4*)(vp2);
            vf.q[1] = *(const uint4*)(vp2 + 16);
            ctxacc[nt] = wmma_bf16(pf.v, vf.v, ctxacc[nt]);
        }
        __syncthreads();  // VT/Vstage free before next iteration
    }

    // ---- epilogue: divide by row sums, store ctx bf16 [tok, d_model] ----
    const float inv = 1.0f / lsum;  // valid for q = qc
    const int b = bh >> 4, h = bh & 15;
#pragma unroll
    for (int r = 0; r < 8; ++r) {
        const float ir = __shfl(inv, r + roff, 32);
        const int srow = q0 + wave * 16 + roff + r;
        const size_t tokr = (size_t)b * SEQ + srow;
#pragma unroll
        for (int nt = 0; nt < 8; ++nt) {
            const int e = h * 128 + nt * 16 + qc;
            ctx[tokr * D_MODEL + e] = f2bf(ctxacc[nt][r] * ir);
        }
    }
}

// ---------------------------------------------------------------------------
// Kernel 5: output projection + residual.  out = x + ctx * Wo^T  (f32 out)
// Same 32x128 wave tile as the QKV GEMM.
// ---------------------------------------------------------------------------
__global__ __launch_bounds__(256) void out_proj_kernel(
    const unsigned short* __restrict__ ctxb, const unsigned short* __restrict__ wo,
    const float* __restrict__ x, float* __restrict__ out) {
    const int lane = threadIdx.x & 31;
    const int wave = threadIdx.x >> 5;
    const int m0 = blockIdx.x * 256 + wave * 32;
    const int n0 = blockIdx.y * 128;

    f32x8 acc0[8], acc1[8];
#pragma unroll
    for (int i = 0; i < 8; ++i) { acc0[i] = zero8(); acc1[i] = zero8(); }

    const int arow = m0 + (lane & 15);
    for (int k = 0; k < D_MODEL; k += 32) {
        const bf16x16 a0 = load_a_frag(ctxb, D_MODEL, arow, k, lane);
        const bf16x16 a1 = load_a_frag(ctxb, D_MODEL, arow + 16, k, lane);
#pragma unroll
        for (int nt = 0; nt < 8; ++nt) {
            const bf16x16 b = load_b_frag(wo, D_MODEL, n0 + nt * 16 + (lane & 15), k, lane);
            acc0[nt] = wmma_bf16(a0, b, acc0[nt]);
            acc1[nt] = wmma_bf16(a1, b, acc1[nt]);
        }
    }

    const int roff = (lane & 16) ? 8 : 0;
#pragma unroll
    for (int half = 0; half < 2; ++half) {
        const int rowbase = m0 + half * 16 + roff;
#pragma unroll
        for (int nt = 0; nt < 8; ++nt) {
            const int e = n0 + nt * 16 + (lane & 15);
#pragma unroll
            for (int r = 0; r < 8; ++r) {
                const size_t idx = (size_t)(rowbase + r) * D_MODEL + e;
                const float vvv = half ? acc1[nt][r] : acc0[nt][r];
                out[idx] = x[idx] + vvv;
            }
        }
    }
}

// ---------------------------------------------------------------------------
// Launch
// ---------------------------------------------------------------------------
extern "C" void kernel_launch(void* const* d_in, const int* in_sizes, int n_in,
                              void* d_out, int out_size, void* d_ws, size_t ws_size,
                              hipStream_t stream) {
    (void)in_sizes; (void)n_in; (void)out_size; (void)ws_size;
    const float* x     = (const float*)d_in[0];
    const float* w_q   = (const float*)d_in[1];
    const float* w_k   = (const float*)d_in[2];
    const float* w_v   = (const float*)d_in[3];
    const float* w_o   = (const float*)d_in[4];
    const float* gamma = (const float*)d_in[5];
    const float* beta  = (const float*)d_in[6];

    char* ws = (char*)d_ws;
    size_t off = 0;
    auto take = [&](size_t bytes) {
        unsigned short* p = (unsigned short*)(ws + off);
        off += bytes;
        return p;
    };
    const size_t act_b = (size_t)TOK * D_MODEL * sizeof(unsigned short);      // 16 MB
    const size_t w_b   = (size_t)D_MODEL * D_MODEL * sizeof(unsigned short);  // 8 MB

    unsigned short* nx   = take(act_b);
    unsigned short* wqb  = take(w_b);
    unsigned short* wkb  = take(w_b);
    unsigned short* wvb  = take(w_b);
    unsigned short* wob  = take(w_b);
    unsigned short* qb   = take(act_b);  // [B,H,S,DK]
    unsigned short* kbuf = take(act_b);
    unsigned short* vbuf = take(act_b);
    unsigned short* ctxb = take(act_b);  // [tok, d_model]

    ln_kernel<<<TOK, 256, 0, stream>>>(x, gamma, beta, nx);
    cast_w_kernel<<<dim3(4096, 4), 256, 0, stream>>>(w_q, w_k, w_v, w_o,
                                                     wqb, wkb, wvb, wob);
    qkv_gemm_kernel<<<dim3(TOK / 256, D_MODEL / 128, 3), 256, 0, stream>>>(
        nx, wqb, wkb, wvb, qb, kbuf, vbuf);
    attn_kernel<<<dim3(BATCH * N_HEADS, SEQ / 128), 256, 0, stream>>>(
        qb, kbuf, vbuf, ctxb);
    out_proj_kernel<<<dim3(TOK / 256, D_MODEL / 128), 256, 0, stream>>>(
        ctxb, wob, x, (float*)d_out);
}
